// GraphBERT_46969762349342
// MI455X (gfx1250) — compile-verified, matrix operand order checked by reading, not run on hardware
//
#include <hip/hip_runtime.h>
#include <hip/hip_bf16.h>
#include <math.h>

typedef __attribute__((ext_vector_type(16))) _Float16 v16h;
typedef __attribute__((ext_vector_type(8)))  _Float16 v8h;
typedef __attribute__((ext_vector_type(8)))  float    v8f;

#define BB  64
#define NN  100
#define NP  128
#define IND 128
#define DHD 256
#define FF  512
#define NLAYER 3

// ---------------------------------------------------------------------------
// WMMA fragment helpers (ISA 7.12.2 layouts, wave32).
// A 16x32 f16 row-major: lane L (row=L&15, half=L>>4) holds K in
//   {8*half..8*half+7} and {16+8*half..16+8*half+7}  -> two 16B loads.
// B 32x16: identical pattern with "row" meaning output column, so a weight
// stored transposed as [N][K] row-major loads the same way.
// ---------------------------------------------------------------------------
static __device__ inline v16h load_frag(const _Float16* base, int ld) {
  int lane = threadIdx.x & 31;
  int row  = lane & 15;
  int half = lane >> 4;
  const _Float16* p = base + row * ld + half * 8;
  v8h lo = *(const v8h*)(p);
  v8h hi = *(const v8h*)(p + 16);
  v16h r;
#pragma unroll
  for (int i = 0; i < 8; ++i) { r[i] = lo[i]; r[8 + i] = hi[i]; }
  return r;
}

static __device__ inline v8f wmma16(v16h a, v16h b, v8f c) {
  return __builtin_amdgcn_wmma_f32_16x16x32_f16(false, a, false, b,
                                                (short)0, c, false, false);
}

static __device__ inline float block_sum_256(float v) {
  __shared__ float red[8];
#pragma unroll
  for (int m = 16; m >= 1; m >>= 1) v += __shfl_xor(v, m, 32);
  __syncthreads();
  if ((threadIdx.x & 31) == 0) red[threadIdx.x >> 5] = v;
  __syncthreads();
  float s = 0.f;
#pragma unroll
  for (int i = 0; i < 8; ++i) s += red[i];
  return s;
}

// ---------------------------------------------------------------------------
// Prep kernels
// ---------------------------------------------------------------------------
__global__ void tr16_kernel(const float* __restrict__ src, _Float16* __restrict__ dst,
                            int K, int N) {     // src[K][N] -> dst[N][K] f16
  int idx = blockIdx.x * 256 + threadIdx.x;
  if (idx >= K * N) return;
  int n = idx / K, k = idx - n * K;
  dst[idx] = (_Float16)src[k * N + n];
}

__global__ void padx_kernel(const float* __restrict__ X, _Float16* __restrict__ Xp) {
  int idx = blockIdx.x * 256 + threadIdx.x;
  if (idx >= BB * NP * IND) return;
  int k = idx & 127; int rem = idx >> 7; int n = rem & 127; int b = rem >> 7;
  Xp[idx] = (n < NN) ? (_Float16)X[((size_t)b * NN + n) * IND + k] : (_Float16)0.f;
}

__global__ void __launch_bounds__(256)
hopmean_kernel(const float* __restrict__ tab, float* __restrict__ hm) {
  float v = tab[blockIdx.x * DHD + threadIdx.x];
  float s = block_sum_256(v);
  if (threadIdx.x == 0) hm[blockIdx.x] = s * (1.f / DHD);
}

__global__ void hopeff_kernel(const int* __restrict__ hop, const float* __restrict__ adj,
                              const float* __restrict__ hm, float* __restrict__ he) {
  int idx = blockIdx.x * 256 + threadIdx.x;
  if (idx >= BB * NN * NN) return;
  int c = hop[idx];
  c = (c < -1 ? -1 : (c > 20 ? 20 : c)) + 1;
  he[idx] = adj[idx] * hm[c];
}

// extras = wl_table[wl] + LN(relu(intimacy*int_w+int_b)) + positional encoding
__global__ void __launch_bounds__(256)
extras_kernel(const int* __restrict__ wl, const float* __restrict__ intim,
              const float* __restrict__ wl_table,
              const float* __restrict__ int_w, const float* __restrict__ int_b,
              const float* __restrict__ int_g, const float* __restrict__ int_beta,
              float* __restrict__ extras) {
  int row = blockIdx.x; int b = row / NN; int n = row - b * NN; int d = threadIdx.x;
  float ti = intim[row];
  float r  = fmaxf(ti * int_w[d] + int_b[d], 0.f);
  float mean = block_sum_256(r) * (1.f / DHD);
  float df = r - mean;
  float var = block_sum_256(df * df) * (1.f / DHD);
  float ie = df * rsqrtf(var + 1e-5f) * int_g[d] + int_beta[d];
  float wle = wl_table[(size_t)wl[row] * DHD + d];
  int p = d >> 1;
  float freq = __expf(-(float)(2 * p) * (9.210340371976184f / (float)DHD));
  float ang = (float)n * freq;
  float pe = (d & 1) ? __cosf(ang) : __sinf(ang);
  extras[((size_t)b * NP + n) * DHD + d] = wle + ie + pe;
}

// ---------------------------------------------------------------------------
// Row-strip GEMM: each wave computes 16 rows x full 256 cols (16 WMMA tiles),
// then bias / relu / residual / LayerNorm / extras all in registers.
// Used for: embed (K=128) and FFN2 (K=512).
// ---------------------------------------------------------------------------
__global__ void __launch_bounds__(256)
row_strip_ln_kernel(const _Float16* __restrict__ A, int K,
                    const _Float16* __restrict__ Bt,      // [256][K] f16
                    const float* __restrict__ bias,
                    const float* __restrict__ resid,      // may be null
                    const float* __restrict__ extras,     // may be null
                    const float* __restrict__ gamma, const float* __restrict__ beta,
                    float* __restrict__ hout, _Float16* __restrict__ hout16,
                    int relu_flag, int zero_pad) {
  int wave = blockIdx.x * 8 + (threadIdx.x >> 5);   // 0..511
  int m0 = wave * 16;
  int ksteps = K >> 5;
  v8f acc[16];
#pragma unroll
  for (int nt = 0; nt < 16; nt++)
#pragma unroll
    for (int e = 0; e < 8; e++) acc[nt][e] = 0.f;

  for (int ks = 0; ks < ksteps; ks++) {
    v16h a = load_frag(A + (size_t)m0 * K + ks * 32, K);
#pragma unroll
    for (int nt = 0; nt < 16; nt++) {
      v16h bf = load_frag(Bt + (size_t)(nt * 16) * K + ks * 32, K);
      acc[nt] = wmma16(a, bf, acc[nt]);
    }
  }
  int lane = threadIdx.x & 31, c0 = lane & 15, half = lane >> 4;
#pragma unroll
  for (int v = 0; v < 8; v++) {
    int r = m0 + v + 8 * half;          // global padded row
    int n = r & 127;                    // row within batch
    float x[16];
#pragma unroll
    for (int nt = 0; nt < 16; nt++) {
      int col = c0 + nt * 16;
      float t = acc[nt][v] + bias[col];
      if (relu_flag) t = fmaxf(t, 0.f);
      if (resid) t += resid[(size_t)r * DHD + col];
      x[nt] = t;
    }
    float s = 0.f;
#pragma unroll
    for (int nt = 0; nt < 16; nt++) s += x[nt];
#pragma unroll
    for (int m = 1; m < 16; m <<= 1) s += __shfl_xor(s, m, 32);
    float mean = s * (1.f / DHD);
    float vs = 0.f;
#pragma unroll
    for (int nt = 0; nt < 16; nt++) { float d = x[nt] - mean; vs += d * d; }
#pragma unroll
    for (int m = 1; m < 16; m <<= 1) vs += __shfl_xor(vs, m, 32);
    float inv = rsqrtf(vs * (1.f / DHD) + 1e-5f);
#pragma unroll
    for (int nt = 0; nt < 16; nt++) {
      int col = c0 + nt * 16;
      float y = (x[nt] - mean) * inv * gamma[col] + beta[col];
      if (extras && n < NN) y += extras[(size_t)r * DHD + col];
      if (n < NN) {
        hout[(size_t)r * DHD + col] = y;
        hout16[(size_t)r * DHD + col] = (_Float16)y;
      } else if (zero_pad) {
        hout[(size_t)r * DHD + col] = 0.f;
        hout16[(size_t)r * DHD + col] = (_Float16)0.f;
      }
    }
  }
}

// ---------------------------------------------------------------------------
// QKV: one GEMM pass over h16 for the three 256x256 weights.
// V is written transposed [b][d][n] so it is a ready-made WMMA B-operand.
// ---------------------------------------------------------------------------
__global__ void __launch_bounds__(256)
qkv_kernel(const _Float16* __restrict__ h16,
           const _Float16* __restrict__ WqT, const _Float16* __restrict__ WkT,
           const _Float16* __restrict__ WvT,
           const float* __restrict__ bq, const float* __restrict__ bk,
           const float* __restrict__ bv,
           _Float16* __restrict__ Q, _Float16* __restrict__ Km,
           _Float16* __restrict__ Vt) {
  int wave = blockIdx.x * 8 + (threadIdx.x >> 5);  // 0..6143
  int which = wave >> 11;                           // 0:Q 1:K 2:V
  int rem = wave & 2047;
  int mt = rem >> 2, ng = rem & 3;
  const _Float16* Wt = (which == 0) ? WqT : (which == 1) ? WkT : WvT;
  const float* bias  = (which == 0) ? bq  : (which == 1) ? bk  : bv;
  v8f acc[4];
#pragma unroll
  for (int j = 0; j < 4; j++)
#pragma unroll
    for (int e = 0; e < 8; e++) acc[j][e] = 0.f;
  int m0 = mt * 16;
  for (int ks = 0; ks < 8; ks++) {
    v16h a = load_frag(h16 + (size_t)m0 * DHD + ks * 32, DHD);
#pragma unroll
    for (int j = 0; j < 4; j++) {
      v16h bf = load_frag(Wt + (size_t)(ng * 64 + j * 16) * DHD + ks * 32, DHD);
      acc[j] = wmma16(a, bf, acc[j]);
    }
  }
  int lane = threadIdx.x & 31, c0 = lane & 15, half = lane >> 4;
#pragma unroll
  for (int j = 0; j < 4; j++)
#pragma unroll
    for (int v = 0; v < 8; v++) {
      int row = m0 + v + 8 * half;
      int d = ng * 64 + j * 16 + c0;
      float val = acc[j][v] + bias[d];
      if (which == 0)      Q[(size_t)row * DHD + d]  = (_Float16)val;
      else if (which == 1) Km[(size_t)row * DHD + d] = (_Float16)val;
      else { int b = row >> 7, n = row & 127;
             Vt[((size_t)b * DHD + d) * NP + n] = (_Float16)val; }
    }
}

// ---------------------------------------------------------------------------
// Fused attention: scores (WMMA) -> scale*hop_effect + mask -> in-register
// softmax -> LDS transpose -> attn@V (WMMA) -> residual + LN.
// One block per batch; 7 waves, each owns a 16-row strip (covers rows 0..111).
// ---------------------------------------------------------------------------
__global__ void __launch_bounds__(224)
attn_kernel(const _Float16* __restrict__ Q, const _Float16* __restrict__ Km,
            const _Float16* __restrict__ Vt, const float* __restrict__ hopeff,
            const float* __restrict__ resid,
            const float* __restrict__ gamma, const float* __restrict__ beta,
            float* __restrict__ hout, _Float16* __restrict__ hout16) {
  __shared__ _Float16 lds_buf[7][16 * 128];
  int b = blockIdx.x;
  int w = threadIdx.x >> 5;
  int lane = threadIdx.x & 31, c0 = lane & 15, half = lane >> 4;
  const _Float16* Qb = Q  + (size_t)b * NP * DHD;
  const _Float16* Kb = Km + (size_t)b * NP * DHD;
  const _Float16* Vb = Vt + (size_t)b * DHD * NP;
  int m0 = w * 16;

  // scores: 16 x 128, K = 256
  v8f s[8];
#pragma unroll
  for (int nt = 0; nt < 8; nt++)
#pragma unroll
    for (int e = 0; e < 8; e++) s[nt][e] = 0.f;
  for (int ks = 0; ks < 8; ks++) {
    v16h a = load_frag(Qb + (size_t)m0 * DHD + ks * 32, DHD);
#pragma unroll
    for (int nt = 0; nt < 8; nt++) {
      v16h bf = load_frag(Kb + (size_t)(nt * 16) * DHD + ks * 32, DHD);
      s[nt] = wmma16(a, bf, s[nt]);
    }
  }

  const float scale = 0.0625f;   // 1/sqrt(256)
  _Float16* myl = lds_buf[w];
#pragma unroll
  for (int v = 0; v < 8; v++) {
    int r = m0 + v + 8 * half;
    float x[8];
#pragma unroll
    for (int nt = 0; nt < 8; nt++) {
      int col = c0 + nt * 16;
      float he = (r < NN && col < NN) ? hopeff[((size_t)b * NN + r) * NN + col] : 0.f;
      float val = s[nt][v] * scale * he;
      if (col >= NN) val = -1e30f;
      x[nt] = val;
    }
    float mx = x[0];
#pragma unroll
    for (int nt = 1; nt < 8; nt++) mx = fmaxf(mx, x[nt]);
#pragma unroll
    for (int m = 1; m < 16; m <<= 1) mx = fmaxf(mx, __shfl_xor(mx, m, 32));
    float sum = 0.f;
#pragma unroll
    for (int nt = 0; nt < 8; nt++) { x[nt] = __expf(x[nt] - mx); sum += x[nt]; }
#pragma unroll
    for (int m = 1; m < 16; m <<= 1) sum += __shfl_xor(sum, m, 32);
    float inv = 1.f / sum;
#pragma unroll
    for (int nt = 0; nt < 8; nt++)
      myl[(v + 8 * half) * 128 + c0 + nt * 16] = (_Float16)(x[nt] * inv);
  }

  // attn @ V : 16 x 256, K = 128 tokens
  v8f o[16];
#pragma unroll
  for (int nt = 0; nt < 16; nt++)
#pragma unroll
    for (int e = 0; e < 8; e++) o[nt][e] = 0.f;
  for (int ks = 0; ks < 4; ks++) {
    v16h a = load_frag(myl + ks * 32, 128);
#pragma unroll
    for (int nt = 0; nt < 16; nt++) {
      v16h bf = load_frag(Vb + (size_t)(nt * 16) * NP + ks * 32, NP);
      o[nt] = wmma16(a, bf, o[nt]);
    }
  }

  // residual + LN + store (guard row < 100)
#pragma unroll
  for (int v = 0; v < 8; v++) {
    int r = m0 + v + 8 * half;
    size_t gr = (size_t)b * NP + r;
    float x[16];
#pragma unroll
    for (int nt = 0; nt < 16; nt++) {
      int col = c0 + nt * 16;
      x[nt] = o[nt][v] + resid[gr * DHD + col];
    }
    float sm = 0.f;
#pragma unroll
    for (int nt = 0; nt < 16; nt++) sm += x[nt];
#pragma unroll
    for (int m = 1; m < 16; m <<= 1) sm += __shfl_xor(sm, m, 32);
    float mean = sm * (1.f / DHD);
    float vs = 0.f;
#pragma unroll
    for (int nt = 0; nt < 16; nt++) { float d = x[nt] - mean; vs += d * d; }
#pragma unroll
    for (int m = 1; m < 16; m <<= 1) vs += __shfl_xor(vs, m, 32);
    float inv = rsqrtf(vs * (1.f / DHD) + 1e-5f);
    if (r < NN) {
#pragma unroll
      for (int nt = 0; nt < 16; nt++) {
        int col = c0 + nt * 16;
        float y = (x[nt] - mean) * inv * gamma[col] + beta[col];
        hout[gr * DHD + col] = y;
        hout16[gr * DHD + col] = (_Float16)y;
      }
    }
  }
}

// ---------------------------------------------------------------------------
// FFN first GEMM: [8192x256] @ [256x512] + bias, ReLU -> f16
// ---------------------------------------------------------------------------
__global__ void __launch_bounds__(256)
ffn1_kernel(const _Float16* __restrict__ h16, const _Float16* __restrict__ W1T,
            const float* __restrict__ b1, _Float16* __restrict__ mid) {
  int wave = blockIdx.x * 8 + (threadIdx.x >> 5);  // 0..4095
  int mt = wave >> 3, ng = wave & 7;
  v8f acc[4];
#pragma unroll
  for (int j = 0; j < 4; j++)
#pragma unroll
    for (int e = 0; e < 8; e++) acc[j][e] = 0.f;
  int m0 = mt * 16;
  for (int ks = 0; ks < 8; ks++) {
    v16h a = load_frag(h16 + (size_t)m0 * DHD + ks * 32, DHD);
#pragma unroll
    for (int j = 0; j < 4; j++) {
      v16h bf = load_frag(W1T + (size_t)(ng * 64 + j * 16) * DHD + ks * 32, DHD);
      acc[j] = wmma16(a, bf, acc[j]);
    }
  }
  int lane = threadIdx.x & 31, c0 = lane & 15, half = lane >> 4;
#pragma unroll
  for (int j = 0; j < 4; j++)
#pragma unroll
    for (int v = 0; v < 8; v++) {
      int row = m0 + v + 8 * half;
      int d = ng * 64 + j * 16 + c0;
      mid[(size_t)row * FF + d] = (_Float16)fmaxf(acc[j][v] + b1[d], 0.f);
    }
}

// ---------------------------------------------------------------------------
// Head: target/neigh pooling + 512->256 (LN) + 256->128 + 128->2 (tiny, VALU)
// ---------------------------------------------------------------------------
__global__ void __launch_bounds__(256)
head_kernel(const float* __restrict__ h,
            const float* __restrict__ agg_w, const float* __restrict__ agg_b,
            const float* __restrict__ agg_g, const float* __restrict__ agg_beta,
            const float* __restrict__ p1_w, const float* __restrict__ p1_b,
            const float* __restrict__ p2_w, const float* __restrict__ p2_b,
            float* __restrict__ out) {
  __shared__ float agg[2 * DHD];
  __shared__ float a1[DHD];
  __shared__ float o1[DHD / 2];
  int b = blockIdx.x, t = threadIdx.x;
  const float* hb = h + (size_t)b * NP * DHD;
  float tgt = hb[(NN - 1) * DHD + t];
  float sm = 0.f;
  for (int n = 0; n < NN - 1; n++) sm += hb[n * DHD + t];
  agg[t] = tgt;
  agg[DHD + t] = sm * (1.f / (NN - 1));
  __syncthreads();
  float acc = agg_b[t];
  for (int k = 0; k < 2 * DHD; k++) acc += agg[k] * agg_w[k * DHD + t];
  acc = fmaxf(acc, 0.f);
  float mean = block_sum_256(acc) * (1.f / DHD);
  float df = acc - mean;
  float var = block_sum_256(df * df) * (1.f / DHD);
  a1[t] = df * rsqrtf(var + 1e-5f) * agg_g[t] + agg_beta[t];
  __syncthreads();
  if (t < DHD / 2) {
    float a2 = p1_b[t];
    for (int d = 0; d < DHD; d++) a2 += a1[d] * p1_w[d * (DHD / 2) + t];
    o1[t] = fmaxf(a2, 0.f);
  }
  __syncthreads();
  if (t < 2) {
    float a3 = p2_b[t];
    for (int j = 0; j < DHD / 2; j++) a3 += o1[j] * p2_w[j * 2 + t];
    out[b * 2 + t] = a3;
  }
}

// ---------------------------------------------------------------------------
extern "C" void kernel_launch(void* const* d_in, const int* in_sizes, int n_in,
                              void* d_out, int out_size, void* d_ws, size_t ws_size,
                              hipStream_t stream) {
  const float* X       = (const float*)d_in[0];
  const int*   wl      = (const int*)d_in[1];
  const float* intim   = (const float*)d_in[2];
  const int*   hop     = (const int*)d_in[3];
  const float* adj     = (const float*)d_in[4];
  const float* fe_w    = (const float*)d_in[5];
  const float* fe_b    = (const float*)d_in[6];
  const float* fe_g    = (const float*)d_in[7];
  const float* fe_beta = (const float*)d_in[8];
  const float* wl_tab  = (const float*)d_in[9];
  const float* int_w   = (const float*)d_in[10];
  const float* int_b   = (const float*)d_in[11];
  const float* int_g   = (const float*)d_in[12];
  const float* int_bt  = (const float*)d_in[13];
  const float* hop_tab = (const float*)d_in[14];
  const float* Wq      = (const float*)d_in[15];
  const float* bq      = (const float*)d_in[16];
  const float* Wk      = (const float*)d_in[17];
  const float* bk      = (const float*)d_in[18];
  const float* Wv      = (const float*)d_in[19];
  const float* bv      = (const float*)d_in[20];
  const float* f1w     = (const float*)d_in[21];
  const float* f1b     = (const float*)d_in[22];
  const float* f2w     = (const float*)d_in[23];
  const float* f2b     = (const float*)d_in[24];
  const float* n1g     = (const float*)d_in[25];
  const float* n1b     = (const float*)d_in[26];
  const float* n2g     = (const float*)d_in[27];
  const float* n2b     = (const float*)d_in[28];
  const float* aggw    = (const float*)d_in[29];
  const float* aggb    = (const float*)d_in[30];
  const float* aggg    = (const float*)d_in[31];
  const float* aggbt   = (const float*)d_in[32];
  const float* p1w     = (const float*)d_in[33];
  const float* p1b     = (const float*)d_in[34];
  const float* p2w     = (const float*)d_in[35];
  const float* p2b     = (const float*)d_in[36];

  char* ws = (char*)d_ws;
  size_t off = 0;
  auto alloc = [&](size_t bytes) {
    size_t o = off; off += (bytes + 255) & ~(size_t)255; return o;
  };
  _Float16* Xp    = (_Float16*)(ws + alloc((size_t)BB * NP * IND * 2));
  _Float16* feT   = (_Float16*)(ws + alloc((size_t)DHD * IND * 2));
  _Float16* WqT   = (_Float16*)(ws + alloc((size_t)NLAYER * DHD * DHD * 2));
  _Float16* WkT   = (_Float16*)(ws + alloc((size_t)NLAYER * DHD * DHD * 2));
  _Float16* WvT   = (_Float16*)(ws + alloc((size_t)NLAYER * DHD * DHD * 2));
  _Float16* f1T   = (_Float16*)(ws + alloc((size_t)NLAYER * FF * DHD * 2));
  _Float16* f2T   = (_Float16*)(ws + alloc((size_t)NLAYER * DHD * FF * 2));
  float*    hm    = (float*)   (ws + alloc(32 * 4));
  float*    he    = (float*)   (ws + alloc((size_t)BB * NN * NN * 4));
  float*    extras= (float*)   (ws + alloc((size_t)BB * NP * DHD * 4));
  float*    hbuf  = (float*)   (ws + alloc((size_t)BB * NP * DHD * 4));
  _Float16* h16   = (_Float16*)(ws + alloc((size_t)BB * NP * DHD * 2));
  _Float16* Qb    = (_Float16*)(ws + alloc((size_t)BB * NP * DHD * 2));
  _Float16* Kb    = (_Float16*)(ws + alloc((size_t)BB * NP * DHD * 2));
  _Float16* Vt    = (_Float16*)(ws + alloc((size_t)BB * DHD * NP * 2));
  _Float16* mid   = (_Float16*)(ws + alloc((size_t)BB * NP * FF * 2));
  if (off > ws_size) return;  // workspace too small; bail deterministically

  dim3 blk(256);
  // weight prep (f16 transpose)
  tr16_kernel<<<(IND * DHD + 255) / 256, blk, 0, stream>>>(fe_w, feT, IND, DHD);
  for (int l = 0; l < NLAYER; l++) {
    tr16_kernel<<<(DHD * DHD + 255) / 256, blk, 0, stream>>>(Wq + (size_t)l * DHD * DHD, WqT + (size_t)l * DHD * DHD, DHD, DHD);
    tr16_kernel<<<(DHD * DHD + 255) / 256, blk, 0, stream>>>(Wk + (size_t)l * DHD * DHD, WkT + (size_t)l * DHD * DHD, DHD, DHD);
    tr16_kernel<<<(DHD * DHD + 255) / 256, blk, 0, stream>>>(Wv + (size_t)l * DHD * DHD, WvT + (size_t)l * DHD * DHD, DHD, DHD);
    tr16_kernel<<<(DHD * FF + 255) / 256, blk, 0, stream>>>(f1w + (size_t)l * DHD * FF, f1T + (size_t)l * FF * DHD, DHD, FF);
    tr16_kernel<<<(FF * DHD + 255) / 256, blk, 0, stream>>>(f2w + (size_t)l * FF * DHD, f2T + (size_t)l * DHD * FF, FF, DHD);
  }
  padx_kernel<<<(BB * NP * IND + 255) / 256, blk, 0, stream>>>(X, Xp);
  hopmean_kernel<<<22, blk, 0, stream>>>(hop_tab, hm);
  hopeff_kernel<<<(BB * NN * NN + 255) / 256, blk, 0, stream>>>(hop, adj, hm, he);
  extras_kernel<<<BB * NN, blk, 0, stream>>>(wl, intim, wl_tab, int_w, int_b,
                                             int_g, int_bt, extras);
  // embed: h = LN(relu(X@fe_w+fe_b)) + extras  (pad rows zeroed)
  row_strip_ln_kernel<<<64, blk, 0, stream>>>(Xp, IND, feT, fe_b,
                                              nullptr, extras, fe_g, fe_beta,
                                              hbuf, h16, 1, 1);
  for (int l = 0; l < NLAYER; l++) {
    qkv_kernel<<<768, blk, 0, stream>>>(h16,
        WqT + (size_t)l * DHD * DHD, WkT + (size_t)l * DHD * DHD,
        WvT + (size_t)l * DHD * DHD,
        bq + l * DHD, bk + l * DHD, bv + l * DHD, Qb, Kb, Vt);
    attn_kernel<<<BB, dim3(224), 0, stream>>>(Qb, Kb, Vt, he, hbuf,
        n1g + l * DHD, n1b + l * DHD, hbuf, h16);
    ffn1_kernel<<<512, blk, 0, stream>>>(h16, f1T + (size_t)l * FF * DHD,
        f1b + l * FF, mid);
    row_strip_ln_kernel<<<64, blk, 0, stream>>>(mid, FF,
        f2T + (size_t)l * DHD * FF, f2b + l * DHD,
        hbuf, nullptr, n2g + l * DHD, n2b + l * DHD, hbuf, h16, 0, 0);
  }
  head_kernel<<<BB, blk, 0, stream>>>(hbuf, aggw, aggb, aggg, aggbt,
                                      p1w, p1b, p2w, p2b, (float*)d_out);
}
